// MLCommons_SSDMobileNetV1_14912126452069
// MI455X (gfx1250) — matrix operand order repaired and checked by reading, not compiled, not run on hardware
//
#include <hip/hip_runtime.h>
#include <hip/hip_bf16.h>
#include <stdint.h>

typedef __attribute__((ext_vector_type(16))) _Float16 v16h;
typedef __attribute__((ext_vector_type(8)))  _Float16 v8h;
typedef __attribute__((ext_vector_type(8)))  float    v8f;

#define NCLS 91
#define NROWS 1917
#define BATCH 32

// ---------------------------------------------------------------------------
// f32 -> f16 weight pre-conversion (run once per launch; ~10 MB total).
// ---------------------------------------------------------------------------
__global__ void cvt_f32_f16_kernel(const float* __restrict__ src,
                                   _Float16* __restrict__ dst, int n)
{
    int i = blockIdx.x * blockDim.x + threadIdx.x;
    if (i < n) dst[i] = (_Float16)src[i];
}

// ---------------------------------------------------------------------------
// Stem: 3x3 s2 conv, Ci=3, NCHW f32 images -> NHWC f16, bias + relu6.
// ---------------------------------------------------------------------------
__global__ void stem_conv_kernel(const float* __restrict__ X,   // [B,3,300,300]
                                 const float* __restrict__ W,   // [32,3,3,3]
                                 const float* __restrict__ bias,
                                 _Float16* __restrict__ Y,      // NHWC [B,150,150,32]
                                 int Hi, int Ho, int stride, int pad, int total)
{
    int idx = blockIdx.x * blockDim.x + threadIdx.x;
    if (idx >= total) return;
    int co = idx & 31; int t = idx >> 5;
    int wo = t % Ho;   t /= Ho;
    int ho = t % Ho;   int b = t / Ho;

    float acc = bias[co];
    int hi0 = ho * stride - pad;
    int wi0 = wo * stride - pad;
    const float* xb = X + (size_t)b * 3 * Hi * Hi;
    const float* wc = W + co * 27;
#pragma unroll
    for (int ci = 0; ci < 3; ++ci) {
        const float* xc = xb + (size_t)ci * Hi * Hi;
        const float* wk = wc + ci * 9;
#pragma unroll
        for (int kh = 0; kh < 3; ++kh) {
            int hi = hi0 + kh;
            if (hi < 0 || hi >= Hi) continue;
#pragma unroll
            for (int kw = 0; kw < 3; ++kw) {
                int wi = wi0 + kw;
                if (wi < 0 || wi >= Hi) continue;
                acc += xc[hi * Hi + wi] * wk[kh * 3 + kw];
            }
        }
    }
    Y[idx] = (_Float16)fminf(fmaxf(acc, 0.0f), 6.0f);
}

// ---------------------------------------------------------------------------
// Depthwise 3x3 + folded BN + relu6, NHWC f16 in/out. Channel-coalesced.
// ---------------------------------------------------------------------------
__global__ void dw3x3_bn_relu6_kernel(const _Float16* __restrict__ X,  // NHWC
                                      const float* __restrict__ W,     // [C,1,3,3]
                                      const float* __restrict__ g,
                                      const float* __restrict__ bta,
                                      _Float16* __restrict__ Y,        // NHWC
                                      int C, int Hi, int Wi, int Ho, int Wo,
                                      int stride, int pad, int total)
{
    int idx = blockIdx.x * blockDim.x + threadIdx.x;
    if (idx >= total) return;
    int c  = idx % C;  int t = idx / C;
    int wo = t % Wo;   t /= Wo;
    int ho = t % Ho;   int b = t / Ho;

    const _Float16* xb = X + (size_t)b * Hi * Wi * C + c;
    const float* wk = W + c * 9;
    float acc = 0.0f;
    int hi0 = ho * stride - pad;
    int wi0 = wo * stride - pad;
#pragma unroll
    for (int kh = 0; kh < 3; ++kh) {
        int hi = hi0 + kh;
        if (hi < 0 || hi >= Hi) continue;
#pragma unroll
        for (int kw = 0; kw < 3; ++kw) {
            int wi = wi0 + kw;
            if (wi < 0 || wi >= Wi) continue;
            acc += (float)xb[((size_t)hi * Wi + wi) * C] * wk[kh * 3 + kw];
        }
    }
    acc = acc * g[c] + bta[c];
    Y[idx] = (_Float16)fminf(fmaxf(acc, 0.0f), 6.0f);
}

// ---------------------------------------------------------------------------
// Direct 3x3 conv (extras second convs), NHWC f16 in/out, bias + relu6.
// ---------------------------------------------------------------------------
__global__ void conv3x3_relu6_nhwc(const _Float16* __restrict__ X,  // NHWC
                                   const float* __restrict__ W,     // OIHW [Co,Ci,3,3]
                                   const float* __restrict__ bias,
                                   _Float16* __restrict__ Y,        // NHWC
                                   int Ci, int Co, int Hi, int Wi,
                                   int Ho, int Wo, int stride, int pad, int total)
{
    int idx = blockIdx.x * blockDim.x + threadIdx.x;
    if (idx >= total) return;
    int co = idx % Co; int t = idx / Co;
    int wo = t % Wo;   t /= Wo;
    int ho = t % Ho;   int b = t / Ho;

    float acc = bias[co];
    int hi0 = ho * stride - pad;
    int wi0 = wo * stride - pad;
    const _Float16* xb = X + (size_t)b * Hi * Wi * Ci;
    const float* wc = W + (size_t)co * Ci * 9;
#pragma unroll
    for (int kh = 0; kh < 3; ++kh) {
        int hi = hi0 + kh;
        if (hi < 0 || hi >= Hi) continue;
#pragma unroll
        for (int kw = 0; kw < 3; ++kw) {
            int wi = wi0 + kw;
            if (wi < 0 || wi >= Wi) continue;
            const _Float16* xp = xb + ((size_t)hi * Wi + wi) * Ci;
            const float* wp = wc + kh * 3 + kw;
            for (int ci = 0; ci < Ci; ++ci)
                acc += (float)xp[ci] * wp[ci * 9];
        }
    }
    Y[idx] = (_Float16)fminf(fmaxf(acc, 0.0f), 6.0f);
}

// ---------------------------------------------------------------------------
// Fragment loaders: pure f16 vector loads, zero converts.
// A row is contiguous [Ci]; a-frag = halfs [ka..ka+7] ++ [ka+8..ka+15].
// B row is contiguous [Ci]; b-frag = halfs [kb..kb+15].
// ---------------------------------------------------------------------------
__device__ __forceinline__ v16h load_a_frag(const _Float16* __restrict__ wrow, int ka)
{
    v8h lo = *(const v8h*)(wrow + ka);
    v8h hi = *(const v8h*)(wrow + ka + 8);
    return __builtin_shufflevector(lo, hi, 0,1,2,3,4,5,6,7,8,9,10,11,12,13,14,15);
}

__device__ __forceinline__ v16h load_b_frag(const _Float16* __restrict__ xrow, int kb)
{
    return *(const v16h*)(xrow + kb);
}

// ---------------------------------------------------------------------------
// Pointwise (1x1) conv as f16 WMMA GEMM, NHWC f16: Y[n,co]=relu6(W[co,:]X[n,:]+b)
// N = B*HW (always /16), Co multiple of 16 -> fully unconditional.
// blockDim = (32,4): 4 N-tiles per block, one wave per 16x16 D tile.
// ---------------------------------------------------------------------------
__global__ void pw_gemm_relu6_wmma(const _Float16* __restrict__ X,  // [N,Ci]
                                   const _Float16* __restrict__ W,  // [Co,Ci] f16
                                   const float* __restrict__ bias,
                                   _Float16* __restrict__ Y,        // [N,Co]
                                   int Ci, int Co)
{
    const int lane = threadIdx.x;
    const int half = lane >> 4;
    const int l16  = lane & 15;
    const int m0   = blockIdx.y * 16;
    const int nt   = blockIdx.x * blockDim.y + threadIdx.y;
    const int n    = nt * 16 + l16;

    const _Float16* xrow = X + (size_t)n * Ci;
    const _Float16* wrow = W + (size_t)(m0 + l16) * Ci;

    v8f acc = {};
    for (int k0 = 0; k0 < Ci; k0 += 32) {
        v16h a = load_a_frag(wrow, k0 + 16 * half);
        v16h b = load_b_frag(xrow, k0 + 16 * half);
        acc = __builtin_amdgcn_wmma_f32_16x16x32_f16(
                  false, a, false, b, (short)0, acc, false, false);
    }

    v8h out;
#pragma unroll
    for (int r = 0; r < 8; ++r) {
        float v = acc[r] + bias[m0 + 8 * half + r];
        out[r] = (_Float16)fminf(fmaxf(v, 0.0f), 6.0f);
    }
    *(v8h*)(Y + (size_t)n * Co + m0 + 8 * half) = out;
}

// ---------------------------------------------------------------------------
// Head 1x1 conv as f16 WMMA GEMM, f16 NHWC input, scattered f32 [B,1917,C] out.
// M = A*C not /16: A-row index clamped (OOB rows feed only unstored D rows).
// ---------------------------------------------------------------------------
__global__ void head_gemm_wmma(const _Float16* __restrict__ X,  // [N,Ci]
                               const _Float16* __restrict__ W,  // [M,Ci] f16
                               const float* __restrict__ bias,
                               float* __restrict__ Out,
                               int Ci, int M, int HW,
                               int A, int C, int aoff)
{
    const int lane = threadIdx.x;
    const int half = lane >> 4;
    const int l16  = lane & 15;
    const int m0   = blockIdx.y * 16;
    const int nt   = blockIdx.x * blockDim.y + threadIdx.y;
    const int n    = nt * 16 + l16;

    const int m_a = m0 + l16;
    const int m_e = (m_a < M) ? m_a : 0;   // clamp: loads stay unconditional
    const _Float16* xrow = X + (size_t)n * Ci;
    const _Float16* wrow = W + (size_t)m_e * Ci;

    v8f acc = {};
    for (int k0 = 0; k0 < Ci; k0 += 32) {
        v16h a = load_a_frag(wrow, k0 + 16 * half);
        v16h b = load_b_frag(xrow, k0 + 16 * half);
        acc = __builtin_amdgcn_wmma_f32_16x16x32_f16(
                  false, a, false, b, (short)0, acc, false, false);
    }

    const int b_img = n / HW;
    const int p     = n % HW;
#pragma unroll
    for (int r = 0; r < 8; ++r) {
        int m = m0 + 8 * half + r;
        if (m < M) {
            float v = acc[r] + bias[m];
            int ai = m / C;
            int ci = m % C;
            size_t row = (size_t)b_img * NROWS + aoff + (size_t)p * A + ai;
            Out[row * C + ci] = v;
        }
    }
}

// ---------------------------------------------------------------------------
// Host orchestration
// ---------------------------------------------------------------------------
static const int DW_CI[13] = {32,64,128,128,256,256,512,512,512,512,512,512,1024};
static const int DW_CO[13] = {64,128,128,256,256,512,512,512,512,512,512,1024,1024};
static const int DW_ST[13] = {1,2,1,2,1,2,1,1,1,1,1,2,1};
static const int EX_CI[4] = {1024,512,256,256};
static const int EX_CM[4] = {256,128,128,64};
static const int EX_CO[4] = {512,256,256,128};
static const int HEAD_CI[6] = {512,1024,512,256,256,128};
static const int HEAD_A[6]  = {3,6,6,6,6,6};
static const int HEAD_H[6]  = {19,10,5,3,2,1};
static const int HEAD_OFF[6]= {0,1083,1683,1833,1887,1911};

static inline int same_pad(int Hi, int Ho, int stride) {
    int tot = (Ho - 1) * stride + 3 - Hi;
    if (tot < 0) tot = 0;
    return tot / 2;
}

extern "C" void kernel_launch(void* const* d_in, const int* in_sizes, int n_in,
                              void* d_out, int out_size, void* d_ws, size_t ws_size,
                              hipStream_t stream) {
    (void)in_sizes; (void)n_in; (void)out_size; (void)ws_size;
    const float* images = (const float*)d_in[0];
    _Float16* hws = (_Float16*)d_ws;

    // workspace layout (_Float16 elements), all activations NHWC f16
    const size_t OFF_A   = 0;          // 46,080,000 (max layer: 150*150*64*32)
    const size_t OFF_B   = 46080000;   // 46,080,000
    const size_t OFF_F1  = 92160000;   // 19*19*512*32
    const size_t OFF_F2  = 98074624;   // 10*10*1024*32
    const size_t OFF_EM  = 101351424;  // extras mid (max 10*10*256*32)
    const size_t OFF_E0  = 102170624;  // 5*5*512*32
    const size_t OFF_E1  = 102580224;  // 3*3*256*32
    const size_t OFF_E2  = 102653952;  // 2*2*256*32
    const size_t OFF_E3  = 102686720;  // 1*1*128*32
    const size_t OFF_W   = 102690816;  // f16 weight staging (~5M halfs)
    _Float16* bufA = hws + OFF_A;
    _Float16* bufB = hws + OFF_B;
    _Float16* f1   = hws + OFF_F1;
    _Float16* f2   = hws + OFF_F2;
    _Float16* emid = hws + OFF_EM;
    _Float16* eout[4] = { hws + OFF_E0, hws + OFF_E1, hws + OFF_E2, hws + OFF_E3 };

    // ---- pre-convert all GEMM weights f32 -> f16 (once per launch) ----
    size_t woff = OFF_W;
    auto cvt = [&](int idx, size_t nelem) -> const _Float16* {
        _Float16* dst = hws + woff;
        int n = (int)nelem;
        cvt_f32_f16_kernel<<<(n + 255) / 256, 256, 0, stream>>>(
            (const float*)d_in[idx], dst, n);
        woff += (nelem + 7) & ~(size_t)7;
        return dst;
    };
    const _Float16* pw_w[13];
    for (int i = 0; i < 13; ++i)
        pw_w[i] = cvt(3 + 5 * i + 3, (size_t)DW_CO[i] * DW_CI[i]);
    const _Float16* ex_w1[4];
    for (int j = 0; j < 4; ++j)
        ex_w1[j] = cvt(68 + 4 * j, (size_t)EX_CM[j] * EX_CI[j]);
    const _Float16* hd_cls[6];
    const _Float16* hd_reg[6];
    for (int h = 0; h < 6; ++h) {
        hd_cls[h] = cvt(84 + 4 * h,     (size_t)HEAD_A[h] * NCLS * HEAD_CI[h]);
        hd_reg[h] = cvt(84 + 4 * h + 2, (size_t)HEAD_A[h] * 4    * HEAD_CI[h]);
    }

    // ---- stem: 3x3 s2, 3->32, 300->150 (TF SAME), NCHW f32 -> NHWC f16 ----
    {
        int Hi = 300, Ho = 150, pad = same_pad(Hi, Ho, 2);
        int total = BATCH * Ho * Ho * 32;
        stem_conv_kernel<<<(total + 255) / 256, 256, 0, stream>>>(
            images, (const float*)d_in[1], (const float*)d_in[2], bufA,
            Hi, Ho, 2, pad, total);
    }

    // ---- 13 depthwise-separable blocks (NHWC f16) ----
    const _Float16* cur = bufA;
    int H = 150;
    for (int i = 0; i < 13; ++i) {
        int Ci = DW_CI[i], Co = DW_CO[i], st = DW_ST[i];
        int Ho = (H + st - 1) / st;
        int pad = same_pad(H, Ho, st);
        int base = 3 + 5 * i;

        // depthwise 3x3 + BN + relu6 -> bufB
        {
            int total = BATCH * Ho * Ho * Ci;
            dw3x3_bn_relu6_kernel<<<(total + 255) / 256, 256, 0, stream>>>(
                cur, (const float*)d_in[base], (const float*)d_in[base + 1],
                (const float*)d_in[base + 2], bufB,
                Ci, H, H, Ho, Ho, st, pad, total);
        }
        // pointwise 1x1 + bias + relu6 (WMMA GEMM, f16)
        _Float16* dst = bufA;
        if (i == 10) dst = f1;   // block-11 feature (19x19x512)
        if (i == 12) dst = f2;   // block-13 feature (10x10x1024)
        {
            int Ntot = BATCH * Ho * Ho;
            int ntiles = Ntot / 16;                  // N always /16 for B=32
            dim3 blk(32, 4);
            dim3 grd((ntiles + 3) / 4, Co / 16);
            pw_gemm_relu6_wmma<<<grd, blk, 0, stream>>>(
                bufB, pw_w[i], (const float*)d_in[base + 4], dst, Ci, Co);
        }
        cur = dst;
        H = Ho;
    }

    // ---- extras: (1x1 relu6 via WMMA) + (3x3 s2 relu6 direct), NHWC f16 ----
    const _Float16* ein = f2;
    int He = 10;
    for (int j = 0; j < 4; ++j) {
        int base = 68 + 4 * j;
        int Ci = EX_CI[j], Cm = EX_CM[j], Co = EX_CO[j];
        {
            int Ntot = BATCH * He * He;
            int ntiles = (Ntot + 15) / 16;
            dim3 blk(32, 4);
            dim3 grd((ntiles + 3) / 4, Cm / 16);
            pw_gemm_relu6_wmma<<<grd, blk, 0, stream>>>(
                ein, ex_w1[j], (const float*)d_in[base + 1], emid, Ci, Cm);
        }
        int Ho = (He + 1) / 2;
        int pad = same_pad(He, Ho, 2);
        {
            int total = BATCH * Ho * Ho * Co;
            conv3x3_relu6_nhwc<<<(total + 255) / 256, 256, 0, stream>>>(
                emid, (const float*)d_in[base + 2], (const float*)d_in[base + 3],
                eout[j], Cm, Co, He, He, Ho, Ho, 2, pad, total);
        }
        ein = eout[j];
        He = Ho;
    }

    // ---- heads: cls + reg 1x1 convs, scattered into concatenated f32 out ----
    float* logits = (float*)d_out;                                // [32,1917,91]
    float* regs   = (float*)d_out + (size_t)BATCH * NROWS * NCLS; // [32,1917,4]
    const _Float16* feats[6] = { f1, f2, eout[0], eout[1], eout[2], eout[3] };
    for (int h = 0; h < 6; ++h) {
        int base = 84 + 4 * h;
        int Ci = HEAD_CI[h], A = HEAD_A[h], Hf = HEAD_H[h];
        int HW = Hf * Hf;
        int Ntot = BATCH * HW;
        int ntiles = Ntot / 16;
        dim3 blk(32, 4);
        {
            int M = A * NCLS;
            dim3 grd((ntiles + 3) / 4, (M + 15) / 16);
            head_gemm_wmma<<<grd, blk, 0, stream>>>(
                feats[h], hd_cls[h], (const float*)d_in[base + 1],
                logits, Ci, M, HW, A, NCLS, HEAD_OFF[h]);
        }
        {
            int M = A * 4;
            dim3 grd((ntiles + 3) / 4, (M + 15) / 16);
            head_gemm_wmma<<<grd, blk, 0, stream>>>(
                feats[h], hd_reg[h], (const float*)d_in[base + 3],
                regs, Ci, M, HW, A, 4, HEAD_OFF[h]);
        }
    }
}
